// MultiheadAttention_9732395893009
// MI455X (gfx1250) — matrix operand order repaired and checked by reading, not compiled, not run on hardware
//
#include <hip/hip_runtime.h>

typedef _Float16 f16;
typedef __attribute__((ext_vector_type(8)))  _Float16 v8h;
typedef __attribute__((ext_vector_type(16))) _Float16 v16h;
typedef __attribute__((ext_vector_type(8)))  float    v8f;

#define TILE  128
#define KSTEP 32
#define LDSP  (KSTEP + 8)   // padded row stride (80B, 16B-aligned)

// ---- CDNA5 async memory->LDS staging (ASYNCcnt tracked, ISA 15.18.3) ----
__device__ __forceinline__ void async_copy16(unsigned lds_addr, const f16* gp) {
  unsigned long long ga = (unsigned long long)(size_t)gp;
  asm volatile("global_load_async_to_lds_b128 %0, %1, off"
               :: "v"(lds_addr), "v"(ga) : "memory");
}
__device__ __forceinline__ void wait_async() {
  asm volatile("s_wait_asynccnt 0x0" ::: "memory");
}
__device__ __forceinline__ unsigned lds_off(const void* p) {
  // generic pointers to LDS carry the LDS byte offset in the low 32 bits
  return (unsigned)(size_t)p;
}

// ---------------------------------------------------------------- convert
__global__ __launch_bounds__(256)
void cvt_f32_f16(const float* __restrict__ src, f16* __restrict__ dst, long long n) {
  long long i = ((long long)blockIdx.x * 256 + threadIdx.x) * 4;
  if (i + 3 < n) {
    float4 f = *(const float4*)(src + i);
    dst[i + 0] = (f16)f.x; dst[i + 1] = (f16)f.y;
    dst[i + 2] = (f16)f.z; dst[i + 3] = (f16)f.w;
  }
}

// ------------------------------------------------- LDS-tiled transpose
// src: [R x C] row-major (f32 or f16), dst: [C x R] row-major f16.
// Coalesced reads AND writes; batched via grid.z.
template<typename TS>
__global__ __launch_bounds__(256)
void transpose_to_f16(const TS* __restrict__ src, f16* __restrict__ dst,
                      int R, int C, long long sBatch) {
  __shared__ f16 tile[32][33];
  const int tx = threadIdx.x & 31, ty = threadIdx.x >> 5;   // 32 x 8
  const long long bb = blockIdx.z;
  const TS* s = src + bb * sBatch;
  f16*      d = dst + bb * sBatch;
  const int c0 = blockIdx.x * 32, r0 = blockIdx.y * 32;
  #pragma unroll
  for (int i = 0; i < 4; ++i) {
    int r = ty + i * 8;
    tile[r][tx] = (f16)s[(long long)(r0 + r) * C + (c0 + tx)];
  }
  __syncthreads();
  #pragma unroll
  for (int i = 0; i < 4; ++i) {
    int r = ty + i * 8;
    d[(long long)(c0 + r) * R + (r0 + tx)] = tile[tx][r];
  }
}

// ---------------------------------------------------------------- GEMM
// C[M,N] = A[M,K] * B^T + epi, where B is given row-major [N,K] so both
// operands stage as contiguous 16B rows into LDS (no in-kernel transpose).
// EPI 0: f16 store, val*scale (+ bias[n] if HAS_BIAS)
// EPI 1: f32 store, val*scale (+ bias[n] if HAS_BIAS)
template<int EPI, bool HAS_BIAS>
__global__ __launch_bounds__(256)
void gemm_f16(const f16* __restrict__ A, long long sA, int lda,
              const f16* __restrict__ B, long long sB, int ldb,
              void* __restrict__ C, long long sC, int ldc,
              const float* __restrict__ bias, float scale, int K)
{
  __shared__ f16 As[2][TILE][LDSP];
  __shared__ f16 Bs[2][TILE][LDSP];

  const int t    = threadIdx.x;
  const int wave = t >> 5, lane = t & 31;
  const int half = lane >> 4, l16 = lane & 15;
  const int wm   = wave >> 2, wn = wave & 3;      // 2 x 4 waves
  const long long bb = blockIdx.z;
  const int blockM = blockIdx.y * TILE;
  const int blockN = blockIdx.x * TILE;

  const f16* Ab = A + bb * sA;
  const f16* Bb = B + bb * sB;

  const int rowT = t >> 2;            // 0..63 (second chunk adds 64)
  const int seg  = (t & 3) * 8;       // 0,8,16,24

  auto stage = [&](int buf, int k0) {
    #pragma unroll
    for (int c = 0; c < 2; ++c) {
      int row = rowT + c * 64;
      const f16* gpa = Ab + (long long)(blockM + row) * lda + (k0 + seg);
      const f16* gpb = Bb + (long long)(blockN + row) * ldb + (k0 + seg);
      __builtin_prefetch(gpa + KSTEP, 0, 1);
      __builtin_prefetch(gpb + KSTEP, 0, 1);
      async_copy16(lds_off(&As[buf][row][seg]), gpa);
      async_copy16(lds_off(&Bs[buf][row][seg]), gpb);
    }
  };

  v8f acc[4][2] = {};

  stage(0, 0);
  wait_async();
  __syncthreads();

  int cur = 0;
  for (int k0 = 0; k0 < K; k0 += KSTEP) {
    const bool more = (k0 + KSTEP) < K;
    if (more) stage(cur ^ 1, k0 + KSTEP);   // async prefetch of next tile

    // ---- fragment loads per ISA 7.12.2 wave32 layouts ----
    v16h afr[4], bfr[2];
    #pragma unroll
    for (int i = 0; i < 4; ++i) {
      int row = wm * 64 + i * 16 + l16;
      v8h lo = *(const v8h*)&As[cur][row][half * 8];
      v8h hi = *(const v8h*)&As[cur][row][16 + half * 8];
      afr[i] = __builtin_shufflevector(lo, hi, 0,1,2,3,4,5,6,7,8,9,10,11,12,13,14,15);
    }
    #pragma unroll
    for (int j = 0; j < 2; ++j) {
      int n = wn * 32 + j * 16 + l16;
      v8h lo = *(const v8h*)&Bs[cur][n][half * 8];
      v8h hi = *(const v8h*)&Bs[cur][n][16 + half * 8];
      bfr[j] = __builtin_shufflevector(lo, hi, 0,1,2,3,4,5,6,7,8,9,10,11,12,13,14,15);
    }
    #pragma unroll
    for (int i = 0; i < 4; ++i)
      #pragma unroll
      for (int j = 0; j < 2; ++j)
        acc[i][j] = __builtin_amdgcn_wmma_f32_16x16x32_f16(
            false, afr[i], false, bfr[j], (short)0, acc[i][j], false, false);

    if (more) wait_async();
    __syncthreads();
    cur ^= 1;
  }

  // ---- epilogue (coalesced; bias hoisted to one load per N-fragment) ----
  #pragma unroll
  for (int j = 0; j < 2; ++j) {
    const int n = blockN + wn * 32 + j * 16 + l16;
    float bval = 0.f;
    if (HAS_BIAS) bval = bias[n];
    #pragma unroll
    for (int i = 0; i < 4; ++i) {
      #pragma unroll
      for (int r = 0; r < 8; ++r) {
        int m = blockM + wm * 64 + i * 16 + half * 8 + r;
        float val = acc[i][j][r] * scale + bval;
        if (EPI == 0)
          ((f16*)C)[bb * sC + (long long)m * ldc + n] = (f16)val;
        else
          ((float*)C)[bb * sC + (long long)m * ldc + n] = val;
      }
    }
  }
}

// ---------------------------------------------------------------- softmax
__global__ __launch_bounds__(256)
void softmax_inplace_f16(f16* __restrict__ P, int S) {
  __shared__ float red[256];
  const int t = threadIdx.x;
  f16* p = P + (long long)blockIdx.x * S;
  float v[8];
  float mx = -3.4e38f;
  #pragma unroll
  for (int i = 0; i < 8; ++i) { v[i] = (float)p[t + i * 256]; mx = fmaxf(mx, v[i]); }
  red[t] = mx; __syncthreads();
  for (int s = 128; s > 0; s >>= 1) { if (t < s) red[t] = fmaxf(red[t], red[t + s]); __syncthreads(); }
  mx = red[0]; __syncthreads();
  float sum = 0.f;
  #pragma unroll
  for (int i = 0; i < 8; ++i) { v[i] = __expf(v[i] - mx); sum += v[i]; }
  red[t] = sum; __syncthreads();
  for (int s = 128; s > 0; s >>= 1) { if (t < s) red[t] += red[t + s]; __syncthreads(); }
  float inv = 1.f / red[0];
  #pragma unroll
  for (int i = 0; i < 8; ++i) p[t + i * 256] = (f16)(v[i] * inv);
}

// ---------------------------------------------------------------- launch
extern "C" void kernel_launch(void* const* d_in, const int* in_sizes, int n_in,
                              void* d_out, int out_size, void* d_ws, size_t ws_size,
                              hipStream_t stream) {
  (void)in_sizes; (void)n_in; (void)out_size; (void)ws_size;
  const int Bb = 4, S = 2048, D = 1024;
  const long long M  = (long long)Bb * S;      // 8192
  const long long nQ = M * D;                  // 8,388,608
  const long long nW = (long long)D * D;

  const float* q  = (const float*)d_in[0];
  const float* k  = (const float*)d_in[1];
  const float* v  = (const float*)d_in[2];
  const float* Wq = (const float*)d_in[3];
  const float* bq = (const float*)d_in[4];
  const float* Wk = (const float*)d_in[5];
  const float* bk = (const float*)d_in[6];
  const float* Wv = (const float*)d_in[7];
  const float* bv = (const float*)d_in[8];
  const float* Wo = (const float*)d_in[9];
  const float* bo = (const float*)d_in[10];

  f16* w = (f16*)d_ws;
  long long off = 0;
  f16* qh  = w + off; off += nQ;               // dead after qp proj -> reuse as vpT
  f16* kh  = w + off; off += nQ;               // dead after kp proj -> reuse as scr
  f16* vh  = w + off; off += nQ;               // dead after vp proj -> reuse as ao
  f16* WqT = w + off; off += nW;
  f16* WkT = w + off; off += nW;
  f16* WvT = w + off; off += nW;
  f16* WoT = w + off; off += nW;
  f16* qp  = w + off; off += nQ;
  f16* kp  = w + off; off += nQ;
  f16* vp  = w + off; off += nQ;
  f16* P   = w + off; off += (long long)Bb * S * S;
  f16* vpT = qh;
  f16* scr = kh;
  f16* ao  = vh;

  dim3 blk(256);

  // f32 -> f16 converts for activations
  cvt_f32_f16<<<dim3((unsigned)(nQ / 1024)), blk, 0, stream>>>(q, qh, nQ);
  cvt_f32_f16<<<dim3((unsigned)(nQ / 1024)), blk, 0, stream>>>(k, kh, nQ);
  cvt_f32_f16<<<dim3((unsigned)(nQ / 1024)), blk, 0, stream>>>(v, vh, nQ);

  // one-time convert+transpose of weights: W[k][n] f32 -> WT[n][k] f16
  dim3 gwt(D / 32, D / 32, 1);
  transpose_to_f16<float><<<gwt, blk, 0, stream>>>(Wq, WqT, D, D, 0);
  transpose_to_f16<float><<<gwt, blk, 0, stream>>>(Wk, WkT, D, D, 0);
  transpose_to_f16<float><<<gwt, blk, 0, stream>>>(Wv, WvT, D, D, 0);
  transpose_to_f16<float><<<gwt, blk, 0, stream>>>(Wo, WoT, D, D, 0);

  // Projections: (8192x1024) @ W + b -> f16
  dim3 gproj(D / TILE, (unsigned)(M / TILE), 1);          // (8, 64, 1)
  gemm_f16<0, true><<<gproj, blk, 0, stream>>>(qh, 0, D, WqT, 0, D, qp, 0, D, bq, 1.f, D);
  gemm_f16<0, true><<<gproj, blk, 0, stream>>>(kh, 0, D, WkT, 0, D, kp, 0, D, bk, 1.f, D);
  gemm_f16<0, true><<<gproj, blk, 0, stream>>>(vh, 0, D, WvT, 0, D, vp, 0, D, bv, 1.f, D);

  // Scores: per batch, qp (SxD) @ kp^T, scale 1/sqrt(1024) -> f16 P
  dim3 gsc(S / TILE, S / TILE, Bb);                       // (16, 16, 4)
  gemm_f16<0, false><<<gsc, blk, 0, stream>>>(qp, (long long)S * D, D,
                                              kp, (long long)S * D, D,
                                              P,  (long long)S * S, S,
                                              nullptr, 0.03125f, D);

  softmax_inplace_f16<<<dim3(Bb * S), blk, 0, stream>>>(P, S);

  // vp (SxD) -> vpT (DxS) per batch, so attn GEMM needs no in-kernel transpose
  dim3 gvt(D / 32, S / 32, Bb);                           // (32, 64, 4)
  transpose_to_f16<f16><<<gvt, blk, 0, stream>>>(vp, vpT, S, D, (long long)S * D);

  // attn_out = P (SxS) @ vp : B operand is vpT [N=D][K=S] -> f16 ao (coalesced)
  dim3 gav(D / TILE, S / TILE, Bb);                       // (8, 16, 4)
  gemm_f16<0, false><<<gav, blk, 0, stream>>>(P,   (long long)S * S, S,
                                              vpT, (long long)S * D, S,
                                              ao,  (long long)S * D, D,
                                              nullptr, 1.f, S);

  // transpose(0,2,1).reshape scramble == batched SxD -> DxS transpose
  transpose_to_f16<f16><<<gvt, blk, 0, stream>>>(ao, scr, S, D, (long long)S * D);

  // Output projection: (8192x1024) @ Wo + bo -> f32 d_out
  gemm_f16<1, true><<<gproj, blk, 0, stream>>>(scr, 0, D, WoT, 0, D,
                                               d_out, 0, D, bo, 1.f, D);
}
